// QLSTMGen195_65481071404182
// MI455X (gfx1250) — matrix operand order repaired and checked by reading, not compiled, not run on hardware
//
#include <hip/hip_runtime.h>

// ---------- types ----------
typedef __attribute__((ext_vector_type(16))) __bf16 v16bf;
typedef __attribute__((ext_vector_type(8)))  __bf16 v8bf;
typedef __attribute__((ext_vector_type(8)))  float  v8f;
typedef __attribute__((ext_vector_type(4)))  int    v4i;

typedef __attribute__((address_space(1))) v4i* gptr_v4i;
typedef __attribute__((address_space(3))) v4i* lptr_v4i;

#define TT 512
#define BB 64
#define DD 1024
#define HH 1024
#define KTOT 2048   // D + H
#define BH   (BB * HH)
#define TBH  ((size_t)TT * BH)

#if __has_builtin(__builtin_amdgcn_global_load_async_to_lds_b128)
#define HAVE_ASYNC_LDS 1
#else
#define HAVE_ASYNC_LDS 0
#endif

// ---------- helpers ----------
__device__ __forceinline__ unsigned short f2bf(float f) {
    union { float f; unsigned u; } v; v.f = f;
    unsigned u = v.u;
    unsigned r = u + 0x7FFFu + ((u >> 16) & 1u);              // round-to-nearest-even
    if ((u & 0x7FFFFFFFu) > 0x7F800000u) r = u | 0x00400000u; // quiet NaN
    return (unsigned short)(r >> 16);
}

__device__ __forceinline__ float sigmoidf(float x) {
    return 1.0f / (1.0f + __expf(-x));
}

__device__ __forceinline__ v16bf cat16(v8bf lo, v8bf hi) {
    return __builtin_shufflevector(lo, hi, 0,1,2,3,4,5,6,7,8,9,10,11,12,13,14,15);
}

// ---------- one-time preprocessing ----------
// WT[n][k] bf16 row-major (n=0..4095, k=0..2047), n = gate*1024 + h-col,
// gate order f,i,g,o. Source W*[k][h] are (D+H) x H fp32 row-major.
__global__ void k_convert_w(const float* __restrict__ Wf, const float* __restrict__ Wi,
                            const float* __restrict__ Wg, const float* __restrict__ Wo,
                            unsigned short* __restrict__ wt) {
    unsigned idx = blockIdx.x * blockDim.x + threadIdx.x;   // 0 .. 4096*2048-1
    unsigned k = idx & (KTOT - 1);
    unsigned n = idx >> 11;
    unsigned gate = n >> 10;
    unsigned col = n & (HH - 1);
    const float* W = (gate == 0) ? Wf : (gate == 1) ? Wi : (gate == 2) ? Wg : Wo;
    wt[idx] = f2bf(W[(size_t)k * HH + col]);
}

__global__ void k_convert_x(const float* __restrict__ x, unsigned short* __restrict__ xbf) {
    unsigned idx = blockIdx.x * blockDim.x + threadIdx.x;   // 0 .. T*B*D-1
    xbf[idx] = f2bf(x[idx]);
}

__global__ void k_init_state(unsigned short* __restrict__ hbf, float* __restrict__ c) {
    unsigned idx = blockIdx.x * blockDim.x + threadIdx.x;   // 0 .. B*H-1
    hbf[idx] = 0;
    c[idx] = 0.0f;
}

// ---------- fused per-step kernel ----------
// 64 blocks x 128 threads (4 waves). mtile = blockIdx>>4 is constant per block,
// so the block's A strip (16 batch rows x 2048 K, bf16 = 64 KB) is shared by all
// 4 waves: stage it once into LDS with async global->LDS b128 copies, then feed
// WMMA A fragments from LDS (XOR-swizzled chunks for conflict-free banking).
// Both the A (LDS) and B (global/L2) operand streams are prefetched one K
// iteration ahead so ds/load waits are covered by the in-flight WMMAs.
// Each wave accumulates the f,i,g,o gate tiles for its (mtile, jtile) via
// 64 x 4 v_wmma_f32_16x16x32_bf16, then applies the LSTM pointwise update
// in-register. h is double-buffered across steps.
__global__ __launch_bounds__(128) void k_step(const unsigned short* __restrict__ xbf,
                                              const unsigned short* __restrict__ hin,
                                              unsigned short* __restrict__ hout,
                                              const unsigned short* __restrict__ wt,
                                              float* __restrict__ c,
                                              const float* __restrict__ bfv,
                                              const float* __restrict__ biv,
                                              const float* __restrict__ bgv,
                                              const float* __restrict__ bov,
                                              float* __restrict__ out,
                                              int t, int last) {
    __shared__ unsigned short As[16 * KTOT];   // 64 KB, chunk-XOR-swizzled

    const int tid  = threadIdx.x;
    const int wave = tid >> 5;
    const int lane = tid & 31;
    const int half = lane >> 4;     // 0: lanes 0-15, 1: lanes 16-31
    const int lr   = lane & 15;
    const int mtile = blockIdx.x >> 4;                 // constant per block
    const int jtile = (blockIdx.x * 4 + wave) & 63;
    const int jcol0 = jtile * 16;

    const __bf16* x0 = (const __bf16*)xbf + (size_t)t * (BB * DD);
    const __bf16* h0 = (const __bf16*)hin;
    const __bf16* w0 = (const __bf16*)wt;

    // ---- stage A strip into LDS: 4096 chunks of 8 bf16 (16B), swizzle cc^row ----
    // LDS elem offset of chunk (row, cc) = row*2048 + ((cc ^ row) << 3)
    {
        const __bf16* xrow = x0 + (size_t)mtile * 16 * DD;
        const __bf16* hrow = h0 + (size_t)mtile * 16 * HH;
        for (int cidx = tid; cidx < 4096; cidx += 128) {
            int row = cidx >> 8;            // 0..15
            int cc  = cidx & 255;           // chunk within row
            int ko  = cc << 3;              // element offset in K
            const __bf16* gp = (ko < DD) ? (xrow + (size_t)row * DD + ko)
                                         : (hrow + (size_t)row * HH + (ko - DD));
            unsigned short* lp = &As[row * KTOT + ((cc ^ row) << 3)];
#if HAVE_ASYNC_LDS
            __builtin_amdgcn_global_load_async_to_lds_b128(
                (gptr_v4i)gp, (lptr_v4i)lp, 0, 0);
#else
            *(v8bf*)lp = *(const v8bf*)gp;
#endif
        }
#if HAVE_ASYNC_LDS
#if __has_builtin(__builtin_amdgcn_s_wait_asynccnt)
        __builtin_amdgcn_s_wait_asynccnt(0);
#else
        asm volatile("s_wait_asynccnt 0x0" ::: "memory");
#endif
#endif
        __syncthreads();
    }

    // B layout (32x16 bf16): lane column N = jcol0+lr, K {kt..kt+7, kt+16..kt+23} (+8 upper half-wave)
    const __bf16* bb[4];
    #pragma unroll
    for (int g = 0; g < 4; ++g)
        bb[g] = w0 + (size_t)(g * HH + jcol0 + lr) * KTOT + half * 8;

    v8f acc[4];
    #pragma unroll
    for (int g = 0; g < 4; ++g) acc[g] = (v8f){};

    // A layout (16x32 bf16): lanes 0-15 K {kt..kt+7, kt+16..kt+23}; 16-31 +8.
    // Lane's row in LDS = lr; chunk index of first 8-elem group = kt/8 + half.
    const unsigned short* arow = &As[lr * KTOT];

    // ---- prologue: preload A (LDS) and B (global) for kt = 0 ----
    v8bf blo[4], bhi[4];
    #pragma unroll
    for (int g = 0; g < 4; ++g) {
        blo[g] = *(const v8bf*)(bb[g]);
        bhi[g] = *(const v8bf*)(bb[g] + 16);
    }
    v8bf alo = *(const v8bf*)(arow + ((half ^ lr) << 3));
    v8bf ahi = *(const v8bf*)(arow + (((half + 2) ^ lr) << 3));

    // ---- K loop: stage current fragments, prefetch kt+32 (A from LDS, B from
    //      global), then 4 WMMAs. Waits land one full iteration after issue. ----
    for (int kt = 0; kt < KTOT; kt += 32) {
        v16bf af  = cat16(alo, ahi);
        v16bf bf0 = cat16(blo[0], bhi[0]);
        v16bf bf1 = cat16(blo[1], bhi[1]);
        v16bf bf2 = cat16(blo[2], bhi[2]);
        v16bf bf3 = cat16(blo[3], bhi[3]);

        int kn = kt + 32;
        if (kn < KTOT) {
            #pragma unroll
            for (int g = 0; g < 4; ++g) {
                blo[g] = *(const v8bf*)(bb[g] + kn);
                bhi[g] = *(const v8bf*)(bb[g] + kn + 16);
            }
            int ccn = (kn >> 3) + half;
            alo = *(const v8bf*)(arow + ((ccn ^ lr) << 3));
            ahi = *(const v8bf*)(arow + (((ccn + 2) ^ lr) << 3));
        }

        acc[0] = __builtin_amdgcn_wmma_f32_16x16x32_bf16(false, af, false, bf0, (short)0, acc[0], false, false);
        acc[1] = __builtin_amdgcn_wmma_f32_16x16x32_bf16(false, af, false, bf1, (short)0, acc[1], false, false);
        acc[2] = __builtin_amdgcn_wmma_f32_16x16x32_bf16(false, af, false, bf2, (short)0, acc[2], false, false);
        acc[3] = __builtin_amdgcn_wmma_f32_16x16x32_bf16(false, af, false, bf3, (short)0, acc[3], false, false);
    }

    // ---- fused LSTM pointwise update ----
    // C/D layout: VGPR r -> batch row = mtile*16 + half*8 + r, col = jcol0 + lr
    const int jcol = jcol0 + lr;
    const float bF = bfv[jcol], bI = biv[jcol], bG = bgv[jcol], bO = bov[jcol];
    const int row0 = mtile * 16 + half * 8;

    #pragma unroll
    for (int r = 0; r < 8; ++r) {
        int bidx = (row0 + r) * HH + jcol;
        float fg = sigmoidf(acc[0][r] + bF);
        float ig = sigmoidf(acc[1][r] + bI);
        float gg = tanhf   (acc[2][r] + bG);
        float og = sigmoidf(acc[3][r] + bO);
        float cn = fg * c[bidx] + ig * gg;
        float h  = og * tanhf(cn);
        c[bidx] = cn;
        hout[bidx] = f2bf(h);
        out[(size_t)t * BH + bidx] = h;
        if (last) {
            out[TBH + bidx] = h;           // hx
            out[TBH + BH + bidx] = cn;     // cx
        }
    }
}

// ---------- launch ----------
extern "C" void kernel_launch(void* const* d_in, const int* in_sizes, int n_in,
                              void* d_out, int out_size, void* d_ws, size_t ws_size,
                              hipStream_t stream) {
    const float* x  = (const float*)d_in[0];
    const float* Wf = (const float*)d_in[1];
    const float* bf = (const float*)d_in[2];
    const float* Wi = (const float*)d_in[3];
    const float* bi = (const float*)d_in[4];
    const float* Wg = (const float*)d_in[5];
    const float* bg = (const float*)d_in[6];
    const float* Wo = (const float*)d_in[7];
    const float* bo = (const float*)d_in[8];
    float* out = (float*)d_out;

    // workspace layout (bytes)
    char* ws = (char*)d_ws;
    unsigned short* WT  = (unsigned short*)(ws);                         // 4096*2048 bf16 = 16 MB
    unsigned short* XBF = (unsigned short*)(ws + (size_t)16777216);      // T*B*D bf16 = 64 MB
    float*          CST = (float*)(ws + (size_t)16777216 + 67108864);    // 64*1024 f32 = 256 KB
    unsigned short* HA  = (unsigned short*)(ws + (size_t)16777216 + 67108864 + 262144);           // 128 KB
    unsigned short* HB  = (unsigned short*)(ws + (size_t)16777216 + 67108864 + 262144 + 131072);  // 128 KB

    // one-time preprocessing (off the recurrent critical path)
    k_convert_w<<<(4096 * 2048) / 256, 256, 0, stream>>>(Wf, Wi, Wg, Wo, WT);
    k_convert_x<<<(TT * BB * DD) / 256, 256, 0, stream>>>(x, XBF);
    k_init_state<<<BH / 256, 256, 0, stream>>>(HA, CST);

    // recurrent loop: one fused kernel per step, h double-buffered
    for (int t = 0; t < TT; ++t) {
        const unsigned short* hin = (t & 1) ? HB : HA;
        unsigned short*      hout = (t & 1) ? HA : HB;
        k_step<<<64, 128, 0, stream>>>(XBF, hin, hout, WT, CST,
                                       bf, bi, bg, bo, out,
                                       t, (t == TT - 1) ? 1 : 0);
    }
}
